// MultiHeadMiniRNNTransformerModel_14104672600789
// MI455X (gfx1250) — compile-verified, matrix operand order checked by reading, not run on hardware
//
#include <hip/hip_runtime.h>
#include <hip/hip_bf16.h>

// ---------------------------------------------------------------------------
// CDNA5 / gfx1250 implementation. wave32, WMMA bf16 (f32 accumulate) for all
// matmuls: transformer GEMMs, fused attention, mini-RNN recurrence, head GEMM.
// Staging is vectorized: b128 global loads -> v_cvt_pk_bf16 -> b64 LDS stores.
// ---------------------------------------------------------------------------

typedef __bf16 v16bf __attribute__((ext_vector_type(16)));
typedef float  v8f   __attribute__((ext_vector_type(8)));

#define WMMA_BF16(a, b, c) \
  __builtin_amdgcn_wmma_f32_16x16x32_bf16(false, (a), false, (b), (short)0, (c), false, false)

#define BSZ 256
#define TSZ 128
#define BT  32768   // B*T

__device__ inline __bf16 tobf(float x) { return (__bf16)x; }

__device__ inline unsigned pack2(float a, float b) {
  union { __bf16 h[2]; unsigned u; } p;
  p.h[0] = (__bf16)a; p.h[1] = (__bf16)b;
  return p.u;
}

__device__ inline v8f v8zero() {
  v8f z = {0.f,0.f,0.f,0.f,0.f,0.f,0.f,0.f};
  return z;
}

// Load a 16x32 (rows x K) bf16 A/B^T fragment from LDS (row-major, leading
// dim `ld` elements; ld*2 must be a multiple of 16 bytes so the 4-dword
// groups merge into ds_load_b128). Layout per CDNA5 ISA 7.12.2.
__device__ inline v16bf load_frag(const __bf16* p0, int row0, int k0, int ld) {
  int lane = threadIdx.x & 31;
  const __bf16* p = p0 + (row0 + (lane & 15)) * ld + k0 + ((lane >> 4) << 3);
  union { v16bf v; unsigned u[8]; } f;
  const unsigned* q = (const unsigned*)p;
  f.u[0] = q[0]; f.u[1] = q[1]; f.u[2] = q[2]; f.u[3] = q[3];
  f.u[4] = q[8]; f.u[5] = q[9]; f.u[6] = q[10]; f.u[7] = q[11];
  return f.v;
}

// ---------------- block reductions ----------------
__device__ inline float blockReduceSum(float v, float* sm) {
  int lane = threadIdx.x & 31, w = threadIdx.x >> 5;
  #pragma unroll
  for (int off = 16; off; off >>= 1) v += __shfl_xor(v, off, 32);
  if (lane == 0) sm[w] = v;
  __syncthreads();
  int nw = blockDim.x >> 5;
  float r = (threadIdx.x < nw) ? sm[threadIdx.x] : 0.f;
  if (w == 0) {
    #pragma unroll
    for (int off = 8; off; off >>= 1) r += __shfl_xor(r, off, 32);
    if (lane == 0) sm[0] = r;
  }
  __syncthreads();
  float out = sm[0];
  __syncthreads();
  return out;
}

__device__ inline float blockReduceMax(float v, float* sm) {
  int lane = threadIdx.x & 31, w = threadIdx.x >> 5;
  #pragma unroll
  for (int off = 16; off; off >>= 1) v = fmaxf(v, __shfl_xor(v, off, 32));
  if (lane == 0) sm[w] = v;
  __syncthreads();
  int nw = blockDim.x >> 5;
  float r = (threadIdx.x < nw) ? sm[threadIdx.x] : -3.4e38f;
  if (w == 0) {
    #pragma unroll
    for (int off = 8; off; off >>= 1) r = fmaxf(r, __shfl_xor(r, off, 32));
    if (lane == 0) sm[0] = r;
  }
  __syncthreads();
  float out = sm[0];
  __syncthreads();
  return out;
}

// ---------------- embedding ----------------
__global__ __launch_bounds__(256) void embed_kernel(
    const int* __restrict__ x, const float* __restrict__ temb,
    const float* __restrict__ pemb, float* __restrict__ tok,
    float* __restrict__ h) {
  int i = blockIdx.x * 256 + threadIdx.x;  // over BT*128
  int e = i & 127;
  int bt = i >> 7;
  int t = bt & (TSZ - 1);
  int token = x[bt];
  float tv = temb[token * 128 + e];
  tok[i] = tv;
  h[i] = tv + pemb[t * 128 + e];
}

// ---------------- generic GEMM: C[M,N] = ep(A[M,K] * B + bias) -------------
// TRANSB=1: W stored (N,K) row-major (y = A * W^T,  einsum 'bte,fe->btf')
// TRANSB=0: W stored (K,N) row-major (y = A * W,    einsum 'bte,ef->btf')
// EP==1: v = (1+*stan_p) * tanh(v).  bscale: scale B row k by bscale[k].
// M multiple of 128; K multiple of 64; N arbitrary.
// Block: 128x64 tile, 8 waves; each wave: 32x32 quadrant = 4 accumulators,
// 8 v_wmma per K-step of 64.
template <int TRANSB, int EP>
__global__ __launch_bounds__(256) void gemm_kernel(
    const float* __restrict__ A, const float* __restrict__ W,
    const float* __restrict__ bias, float* __restrict__ C,
    int M, int N, int K,
    const float* __restrict__ stan_p, const float* __restrict__ bscale) {
  __shared__ __bf16 As[128 * 72];
  __shared__ __bf16 Bs[64 * 72];
  const int LD = 72;
  int m0 = blockIdx.y * 128, n0 = blockIdx.x * 64;
  int tid = threadIdx.x;
  int wave = tid >> 5, lane = tid & 31, hf = lane >> 4, ln = lane & 15;
  int mb = (wave >> 1) * 32;   // wave's 32-row band within the 128-row tile
  int nb = (wave & 1) * 32;    // wave's 32-col band within the 64-col tile
  v8f acc00 = v8zero(), acc01 = v8zero(), acc10 = v8zero(), acc11 = v8zero();

  for (int k0 = 0; k0 < K; k0 += 64) {
    // stage A: 128x64 f32 -> bf16 (2048 float4 quads, 8 per thread)
    #pragma unroll
    for (int j = 0; j < 8; ++j) {
      int qq = tid + j * 256;
      int r = qq >> 4, qc = (qq & 15) * 4;
      float4 va = *(const float4*)(A + (size_t)(m0 + r) * K + k0 + qc);
      uint2 pk; pk.x = pack2(va.x, va.y); pk.y = pack2(va.z, va.w);
      *(uint2*)(As + r * LD + qc) = pk;
    }
    // stage B^T: Bs[n][k], 64x64
    if (TRANSB) {
      #pragma unroll
      for (int j = 0; j < 4; ++j) {
        int qq = tid + j * 256;
        int r = qq >> 4, qc = (qq & 15) * 4;
        int n = n0 + r;
        float4 vb = {0.f, 0.f, 0.f, 0.f};
        if (n < N) vb = *(const float4*)(W + (size_t)n * K + k0 + qc);
        uint2 pk; pk.x = pack2(vb.x, vb.y); pk.y = pack2(vb.z, vb.w);
        *(uint2*)(Bs + r * LD + qc) = pk;
      }
    } else {
      #pragma unroll
      for (int j = 0; j < 16; ++j) {
        int i = tid + j * 256;
        int r = i >> 6, c = i & 63;
        int n = n0 + r;
        float v = 0.f;
        if (n < N) {
          v = W[(size_t)(k0 + c) * N + n];
          if (bscale) v *= bscale[k0 + c];
        }
        Bs[r * LD + c] = tobf(v);
      }
    }
    // prefetch next A tile into GL2 while we compute
    if (k0 + 64 < K)
      __builtin_prefetch(A + (size_t)(m0 + (tid >> 1)) * K + k0 + 64 + (tid & 1) * 32, 0, 3);
    __syncthreads();
    #pragma unroll
    for (int ks = 0; ks < 64; ks += 32) {
      v16bf a0 = load_frag(As, mb, ks, LD);
      v16bf a1 = load_frag(As, mb + 16, ks, LD);
      v16bf b0 = load_frag(Bs, nb, ks, LD);
      v16bf b1 = load_frag(Bs, nb + 16, ks, LD);
      acc00 = WMMA_BF16(a0, b0, acc00);
      acc01 = WMMA_BF16(a0, b1, acc01);
      acc10 = WMMA_BF16(a1, b0, acc10);
      acc11 = WMMA_BF16(a1, b1, acc11);
    }
    __syncthreads();
  }

  float s = stan_p ? (1.0f + *stan_p) : 0.f;
  v8f accs[2][2] = {{acc00, acc01}, {acc10, acc11}};
  #pragma unroll
  for (int mi = 0; mi < 2; ++mi)
    #pragma unroll
    for (int ni = 0; ni < 2; ++ni)
      #pragma unroll
      for (int r = 0; r < 8; ++r) {
        int row = m0 + mb + mi * 16 + r + hf * 8;
        int col = n0 + nb + ni * 16 + ln;
        if (col < N) {
          float v = accs[mi][ni][r] + (bias ? bias[col] : 0.f);
          if (EP == 1) v = s * tanhf(v);
          C[(size_t)row * N + col] = v;
        }
      }
}

// ---------------- fused non-causal attention, one block per (b, head) ------
struct QKsh { __bf16 q[128 * 72]; __bf16 k[128 * 72]; };
union AttnSh { QKsh qk; __bf16 p[128 * 136]; };

template <int ED, int HD>
__global__ __launch_bounds__(256) void attn_kernel(
    const float* __restrict__ qkv, float* __restrict__ O, float scale) {
  __shared__ AttnSh sh;
  __shared__ __bf16 vt[64 * 136];   // V^T: [d][t]
  int b = blockIdx.x >> 2;
  int h = blockIdx.x & 3;
  int tid = threadIdx.x;
  size_t base = (size_t)b * TSZ * (3 * ED);

  constexpr int QR = HD / 4;             // float4 quads per row
  constexpr int QSH = (HD == 32) ? 3 : 4;
  constexpr int ITER = (128 * QR) / 256; // 4 or 8
  #pragma unroll
  for (int j = 0; j < ITER; ++j) {
    int qq = tid + j * 256;
    int t = qq >> QSH, d4 = (qq & (QR - 1)) * 4;
    size_t off = base + (size_t)t * (3 * ED) + h * HD + d4;
    float4 vq = *(const float4*)(qkv + off);
    float4 vk = *(const float4*)(qkv + off + ED);
    float4 vv = *(const float4*)(qkv + off + 2 * ED);
    uint2 pq; pq.x = pack2(vq.x, vq.y); pq.y = pack2(vq.z, vq.w);
    uint2 pk; pk.x = pack2(vk.x, vk.y); pk.y = pack2(vk.z, vk.w);
    *(uint2*)(sh.qk.q + t * 72 + d4) = pq;
    *(uint2*)(sh.qk.k + t * 72 + d4) = pk;
    vt[(d4 + 0) * 136 + t] = tobf(vv.x);
    vt[(d4 + 1) * 136 + t] = tobf(vv.y);
    vt[(d4 + 2) * 136 + t] = tobf(vv.z);
    vt[(d4 + 3) * 136 + t] = tobf(vv.w);
  }
  __syncthreads();

  int wave = tid >> 5, lane = tid & 31, hf = lane >> 4, ln = lane & 15;
  v8f sacc[8];
  #pragma unroll
  for (int tn = 0; tn < 8; ++tn) sacc[tn] = v8zero();
  #pragma unroll
  for (int k0 = 0; k0 < HD; k0 += 32) {
    v16bf a = load_frag(sh.qk.q, wave * 16, k0, 72);
    #pragma unroll
    for (int tn = 0; tn < 8; ++tn) {
      v16bf bfrag = load_frag(sh.qk.k, tn * 16, k0, 72);
      sacc[tn] = WMMA_BF16(a, bfrag, sacc[tn]);
    }
  }
  __syncthreads();   // done reading q/k; p may alias them now

  // softmax over 128 cols per row, in registers + 16-lane shuffles
  #pragma unroll
  for (int r = 0; r < 8; ++r) {
    float mx = -3.4e38f;
    #pragma unroll
    for (int tn = 0; tn < 8; ++tn) mx = fmaxf(mx, sacc[tn][r]);
    #pragma unroll
    for (int off = 1; off < 16; off <<= 1) mx = fmaxf(mx, __shfl_xor(mx, off, 32));
    float e8[8]; float ssum = 0.f;
    #pragma unroll
    for (int tn = 0; tn < 8; ++tn) {
      e8[tn] = __expf((sacc[tn][r] - mx) * scale);
      ssum += e8[tn];
    }
    #pragma unroll
    for (int off = 1; off < 16; off <<= 1) ssum += __shfl_xor(ssum, off, 32);
    float inv = 1.f / ssum;
    int row = wave * 16 + r + hf * 8;
    #pragma unroll
    for (int tn = 0; tn < 8; ++tn)
      sh.p[row * 136 + tn * 16 + ln] = tobf(e8[tn] * inv);
  }
  __syncthreads();

  constexpr int NDT = HD >> 4;
  v8f oacc[NDT];
  #pragma unroll
  for (int dt = 0; dt < NDT; ++dt) oacc[dt] = v8zero();
  #pragma unroll
  for (int k0 = 0; k0 < 128; k0 += 32) {
    v16bf a = load_frag(sh.p, wave * 16, k0, 136);
    #pragma unroll
    for (int dt = 0; dt < NDT; ++dt) {
      v16bf bfrag = load_frag(vt, dt * 16, k0, 136);
      oacc[dt] = WMMA_BF16(a, bfrag, oacc[dt]);
    }
  }
  #pragma unroll
  for (int dt = 0; dt < NDT; ++dt)
    #pragma unroll
    for (int r = 0; r < 8; ++r) {
      int row = wave * 16 + r + hf * 8;
      int col = dt * 16 + ln;
      O[((size_t)b * TSZ + row) * ED + h * HD + col] = oacc[dt][r];
    }
}

// ---------------- residual + layernorm: x = LN(x + rs*o) -------------------
__global__ __launch_bounds__(256) void resln_kernel(
    float* __restrict__ x, const float* __restrict__ o, float rs,
    const float* __restrict__ g, const float* __restrict__ bb, int Ed) {
  __shared__ float sm[8];
  int row = blockIdx.x;
  int tid = threadIdx.x;
  float v = 0.f;
  if (tid < Ed) v = x[(size_t)row * Ed + tid] + rs * o[(size_t)row * Ed + tid];
  float m = blockReduceSum(v, sm) / (float)Ed;
  float c = (tid < Ed) ? (v - m) : 0.f;
  float var = blockReduceSum(c * c, sm) / (float)Ed;
  if (tid < Ed)
    x[(size_t)row * Ed + tid] = c * rsqrtf(var + 1e-5f) * g[tid] + bb[tid];
}

// ---------------- mini-RNN: one block per (head, 64-batch chunk) -----------
__global__ __launch_bounds__(256) void rnn_kernel(
    const float* __restrict__ tok, const float* __restrict__ h1,
    const float* __restrict__ win, const float* __restrict__ binb,
    const float* __restrict__ wh, const float* __restrict__ bhb,
    const float* __restrict__ stan, const float* __restrict__ wx,
    const float* __restrict__ bxb, float* __restrict__ h2) {
  __shared__ __bf16 winT[64 * 72], whT[64 * 72], wxT[64 * 40];
  __shared__ __bf16 combS[64 * 72], hprevS[64 * 72], xorS[64 * 40];
  __shared__ float hnf[64 * 66];
  int h = blockIdx.x & 3;
  int b0 = (blockIdx.x >> 2) * 64;
  int tid = threadIdx.x;

  #pragma unroll
  for (int j = 0; j < 16; ++j) {       // W^T: [o][i], 64x64 each
    int i = tid + j * 256;
    int o = i >> 6, ii = i & 63;
    winT[o * 72 + ii] = tobf(win[((size_t)h * 64 + ii) * 64 + o]);
    whT[o * 72 + ii]  = tobf(wh[((size_t)h * 64 + ii) * 64 + o]);
  }
  #pragma unroll
  for (int j = 0; j < 8; ++j) {        // wx^T: [o][i], 64x32
    int i = tid + j * 256;
    int o = i >> 5, ii = i & 31;
    wxT[o * 40 + ii] = tobf(wx[((size_t)h * 32 + ii) * 64 + o]);
  }
  #pragma unroll
  for (int j = 0; j < 9; ++j)          // zero hprev (64*72 bf16 = 2304 dwords)
    ((unsigned*)hprevS)[tid + j * 256] = 0u;
  float s1 = 1.0f + stan[h];

  int wave = tid >> 5, lane = tid & 31, hf = lane >> 4, ln = lane & 15;
  int t0 = wave * 2, t1 = wave * 2 + 1;
  int tm0 = (t0 >> 2) * 16, tn0 = (t0 & 3) * 16;
  int tm1 = (t1 >> 2) * 16, tn1 = (t1 & 3) * 16;
  __syncthreads();

  for (int t = 0; t < TSZ; ++t) {
    #pragma unroll
    for (int j = 0; j < 4; ++j) {      // comb: 64 rows x 16 quads
      int qq = tid + j * 256;
      int r = qq >> 4, qi = qq & 15;
      size_t bt = ((size_t)(b0 + r) * TSZ + t) * 128;
      const float* src = (qi < 8) ? (tok + bt + h * 32 + qi * 4)
                                  : (h1 + bt + h * 32 + (qi - 8) * 4);
      float4 v = *(const float4*)src;
      uint2 pk; pk.x = pack2(v.x, v.y); pk.y = pack2(v.z, v.w);
      *(uint2*)(combS + r * 72 + qi * 4) = pk;
    }
    __syncthreads();

    v8f a0 = v8zero(), a1 = v8zero();
    #pragma unroll
    for (int k0 = 0; k0 < 64; k0 += 32) {
      v16bf ac = load_frag(combS, tm0, k0, 72);
      v16bf bw = load_frag(winT, tn0, k0, 72);
      a0 = WMMA_BF16(ac, bw, a0);
      v16bf ah = load_frag(hprevS, tm0, k0, 72);
      v16bf bh2 = load_frag(whT, tn0, k0, 72);
      a0 = WMMA_BF16(ah, bh2, a0);
      ac = load_frag(combS, tm1, k0, 72);
      bw = load_frag(winT, tn1, k0, 72);
      a1 = WMMA_BF16(ac, bw, a1);
      ah = load_frag(hprevS, tm1, k0, 72);
      bh2 = load_frag(whT, tn1, k0, 72);
      a1 = WMMA_BF16(ah, bh2, a1);
    }
    #pragma unroll
    for (int r = 0; r < 8; ++r) {
      int row = tm0 + r + hf * 8, col = tn0 + ln;
      hnf[row * 66 + col] = s1 * tanhf(a0[r] + binb[h * 64 + col] + bhb[h * 64 + col]);
      row = tm1 + r + hf * 8; col = tn1 + ln;
      hnf[row * 66 + col] = s1 * tanhf(a1[r] + binb[h * 64 + col] + bhb[h * 64 + col]);
    }
    __syncthreads();

    #pragma unroll
    for (int j = 0; j < 8; ++j) {      // xor gate: 64x32
      int i = tid + j * 256;
      int r = i >> 5, c = i & 31;
      float av = 1.f / (1.f + __expf(-hnf[r * 66 + c]));
      float bv = 1.f / (1.f + __expf(-hnf[r * 66 + c + 32]));
      xorS[r * 40 + c] = tobf(0.5f * (av + bv - 2.f * av * bv));
    }
    __syncthreads();

    v8f x0 = v8zero(), x1 = v8zero();
    {
      v16bf ax = load_frag(xorS, tm0, 0, 40);
      v16bf bw = load_frag(wxT, tn0, 0, 40);
      x0 = WMMA_BF16(ax, bw, x0);
      ax = load_frag(xorS, tm1, 0, 40);
      bw = load_frag(wxT, tn1, 0, 40);
      x1 = WMMA_BF16(ax, bw, x1);
    }
    #pragma unroll
    for (int r = 0; r < 8; ++r) {
      int row = tm0 + r + hf * 8, col = tn0 + ln;
      float hv = hnf[row * 66 + col] + x0[r] + bxb[h * 64 + col];
      h2[((size_t)(b0 + row) * TSZ + t) * 256 + h * 64 + col] = hv;
      hprevS[row * 72 + col] = tobf(hv);
      row = tm1 + r + hf * 8; col = tn1 + ln;
      hv = hnf[row * 66 + col] + x1[r] + bxb[h * 64 + col];
      h2[((size_t)(b0 + row) * TSZ + t) * 256 + h * 64 + col] = hv;
      hprevS[row * 72 + col] = tobf(hv);
    }
    __syncthreads();
  }
}

// ---------------- harmonic head pieces ----------------
__global__ __launch_bounds__(256) void var_kernel(
    const float* __restrict__ h2, float* __restrict__ invv) {
  __shared__ float sm[8];
  int d = blockIdx.x;  // 0..255
  float s = 0.f, ss = 0.f;
  for (int i = threadIdx.x; i < BT; i += 256) {
    float v = h2[(size_t)i * 256 + d];
    s += v; ss += v * v;
  }
  float ts = blockReduceSum(s, sm);
  float tss = blockReduceSum(ss, sm);
  if (threadIdx.x == 0) {
    float n = (float)BT;
    float var = (tss - ts * ts / n) / (n - 1.f) + 1e-8f;  // ddof=1
    invv[d] = 1.f / var;
  }
}

__global__ __launch_bounds__(256) void x2_kernel(
    const float* __restrict__ h2, const float* __restrict__ invv,
    float* __restrict__ x2o) {
  __shared__ float sm[8];
  int row = blockIdx.x;
  float v = h2[(size_t)row * 256 + threadIdx.x];
  float t = blockReduceSum(v * v * invv[threadIdx.x], sm);
  if (threadIdx.x == 0) x2o[row] = t;
}

__global__ __launch_bounds__(256) void w2_kernel(
    const float* __restrict__ unemb, const float* __restrict__ invv,
    float* __restrict__ w2o) {
  __shared__ float sm[8];
  int v = blockIdx.x;        // 0..64
  int d = threadIdx.x;       // 0..255
  float u = unemb[(size_t)d * 65 + v];
  float t = blockReduceSum(u * u * invv[d], sm);
  if (threadIdx.x == 0) w2o[v] = t;
}

__global__ __launch_bounds__(64) void zero_kernel(float* p, int n) {
  int i = blockIdx.x * 64 + threadIdx.x;
  if (i < n) p[i] = 0.f;
}

__global__ __launch_bounds__(256) void dist_kernel(
    const float* __restrict__ cross, const float* __restrict__ x2,
    const float* __restrict__ w2o, float* __restrict__ dbuf,
    float* __restrict__ dsum) {
  __shared__ float sm[8];
  size_t i = (size_t)blockIdx.x * 256 + threadIdx.x;
  float dv = 0.f;
  if (i < (size_t)BT * 65) {
    int row = (int)(i / 65), v = (int)(i % 65);
    float t = x2[row] - 2.f * cross[i] + w2o[v];
    dv = sqrtf(fmaxf(t, 0.f));
    dbuf[i] = dv;
  }
  float bs = blockReduceSum(dv, sm);
  if (threadIdx.x == 0) atomicAdd(dsum, bs);
}

__global__ __launch_bounds__(128) void headsm_kernel(
    const float* __restrict__ dbuf, const float* __restrict__ dsum,
    float* __restrict__ out) {
  __shared__ float sm[8];
  int row = blockIdx.x;
  float scale = fmaxf(dsum[0] * (1.0f / ((float)BT * 65.f)), 1e-8f);
  int v = threadIdx.x;
  float logit = -3.4e38f;
  if (v < 65) {
    float dsv = fmaxf(dbuf[(size_t)row * 65 + v] / scale, 1e-6f);
    logit = -16.f * __logf(dsv + 1e-8f);   // n = sqrt(D) = 16
  }
  float mx = blockReduceMax(logit, sm);
  float e = (v < 65) ? __expf(logit - mx) : 0.f;
  float s = blockReduceSum(e, sm);
  if (v < 65) out[(size_t)row * 65 + v] = e / s;
}

// ---------------------------------------------------------------------------
extern "C" void kernel_launch(void* const* d_in, const int* in_sizes, int n_in,
                              void* d_out, int out_size, void* d_ws, size_t ws_size,
                              hipStream_t stream) {
  (void)in_sizes; (void)n_in; (void)out_size; (void)ws_size;
  const int*   xin  = (const int*)d_in[0];
  const float* temb = (const float*)d_in[1];
  const float* pemb = (const float*)d_in[2];
  // tb1 (L=4, Ed=128, f=512)
  const float* t1_qkv_w = (const float*)d_in[3];
  const float* t1_qkv_b = (const float*)d_in[4];
  const float* t1_out_w = (const float*)d_in[5];
  const float* t1_out_b = (const float*)d_in[6];
  const float* t1_ln1g  = (const float*)d_in[7];
  const float* t1_ln1b  = (const float*)d_in[8];
  const float* t1_ln2g  = (const float*)d_in[9];
  const float* t1_ln2b  = (const float*)d_in[10];
  const float* t1_w1    = (const float*)d_in[11];
  const float* t1_b1    = (const float*)d_in[12];
  const float* t1_stan  = (const float*)d_in[13];
  const float* t1_w2    = (const float*)d_in[14];
  const float* t1_b2    = (const float*)d_in[15];
  // rnn
  const float* r_win  = (const float*)d_in[16];
  const float* r_bin  = (const float*)d_in[17];
  const float* r_wh   = (const float*)d_in[18];
  const float* r_bh   = (const float*)d_in[19];
  const float* r_stan = (const float*)d_in[20];
  const float* r_wx   = (const float*)d_in[21];
  const float* r_bx   = (const float*)d_in[22];
  // tb2 (L=2, Ed=256, f=1024)
  const float* t2_qkv_w = (const float*)d_in[23];
  const float* t2_qkv_b = (const float*)d_in[24];
  const float* t2_out_w = (const float*)d_in[25];
  const float* t2_out_b = (const float*)d_in[26];
  const float* t2_ln1g  = (const float*)d_in[27];
  const float* t2_ln1b  = (const float*)d_in[28];
  const float* t2_ln2g  = (const float*)d_in[29];
  const float* t2_ln2b  = (const float*)d_in[30];
  const float* t2_w1    = (const float*)d_in[31];
  const float* t2_b1    = (const float*)d_in[32];
  const float* t2_stan  = (const float*)d_in[33];
  const float* t2_w2    = (const float*)d_in[34];
  const float* t2_b2    = (const float*)d_in[35];
  const float* unemb    = (const float*)d_in[36];
  float* out = (float*)d_out;

  // workspace layout (floats)
  float* Wp   = (float*)d_ws;
  float* tok  = Wp;                         // BT*128
  float* hA   = tok + (size_t)BT * 128;     // BT*128 (tb1 activations)
  float* hB   = hA + (size_t)BT * 128;      // BT*256 (h2)
  float* big  = hB + (size_t)BT * 256;      // BT*1024 (qkv / mlp-hidden / cross)
  float* obuf = big + (size_t)BT * 1024;    // BT*256
  float* dbuf = obuf + (size_t)BT * 256;    // BT*65
  float* x2b  = dbuf + (size_t)BT * 65;     // BT
  float* invb = x2b + BT;                   // 256
  float* w2b  = invb + 256;                 // 128 (65 used)
  float* dsum = w2b + 128;                  // 1

  // 1) embeddings
  embed_kernel<<<(BT * 128) / 256, 256, 0, stream>>>(xin, temb, pemb, tok, hA);

  // 2) tb1: 4 layers, Ed=128, f=512, hd=32, res_scale = 1/sqrt(4)
  for (int i = 0; i < 4; ++i) {
    const int Ed = 128, F = 512;
    const float rs = 0.5f;
    gemm_kernel<1, 0><<<dim3((3 * Ed) / 64, BT / 128), 256, 0, stream>>>(
        hA, t1_qkv_w + (size_t)i * 3 * Ed * Ed, t1_qkv_b + (size_t)i * 3 * Ed,
        big, BT, 3 * Ed, Ed, nullptr, nullptr);
    attn_kernel<128, 32><<<BSZ * 4, 256, 0, stream>>>(big, obuf, 0.1767766953f);
    gemm_kernel<1, 0><<<dim3(Ed / 64, BT / 128), 256, 0, stream>>>(
        obuf, t1_out_w + (size_t)i * Ed * Ed, t1_out_b + (size_t)i * Ed,
        big, BT, Ed, Ed, nullptr, nullptr);
    resln_kernel<<<BT, 256, 0, stream>>>(hA, big, rs, t1_ln1g + (size_t)i * Ed,
                                         t1_ln1b + (size_t)i * Ed, Ed);
    gemm_kernel<0, 1><<<dim3(F / 64, BT / 128), 256, 0, stream>>>(
        hA, t1_w1 + (size_t)i * Ed * F, t1_b1 + (size_t)i * F,
        big, BT, F, Ed, t1_stan + i, nullptr);
    gemm_kernel<0, 0><<<dim3(Ed / 64, BT / 128), 256, 0, stream>>>(
        big, t1_w2 + (size_t)i * F * Ed, t1_b2 + (size_t)i * Ed,
        obuf, BT, Ed, F, nullptr, nullptr);
    resln_kernel<<<BT, 256, 0, stream>>>(hA, obuf, rs, t1_ln2g + (size_t)i * Ed,
                                         t1_ln2b + (size_t)i * Ed, Ed);
  }

  // 3) per-head mini-RNN over time -> hB (B,T,256)
  rnn_kernel<<<16, 256, 0, stream>>>(tok, hA, r_win, r_bin, r_wh, r_bh,
                                     r_stan, r_wx, r_bx, hB);

  // 4) tb2: 2 layers, Ed=256, f=1024, hd=64, res_scale = 1/sqrt(2)
  for (int i = 0; i < 2; ++i) {
    const int Ed = 256, F = 1024;
    const float rs = 0.7071067812f;
    gemm_kernel<1, 0><<<dim3((3 * Ed) / 64, BT / 128), 256, 0, stream>>>(
        hB, t2_qkv_w + (size_t)i * 3 * Ed * Ed, t2_qkv_b + (size_t)i * 3 * Ed,
        big, BT, 3 * Ed, Ed, nullptr, nullptr);
    attn_kernel<256, 64><<<BSZ * 4, 256, 0, stream>>>(big, obuf, 0.125f);
    gemm_kernel<1, 0><<<dim3(Ed / 64, BT / 128), 256, 0, stream>>>(
        obuf, t2_out_w + (size_t)i * Ed * Ed, t2_out_b + (size_t)i * Ed,
        big, BT, Ed, Ed, nullptr, nullptr);
    resln_kernel<<<BT, 256, 0, stream>>>(hB, big, rs, t2_ln1g + (size_t)i * Ed,
                                         t2_ln1b + (size_t)i * Ed, Ed);
    gemm_kernel<0, 1><<<dim3(F / 64, BT / 128), 256, 0, stream>>>(
        hB, t2_w1 + (size_t)i * Ed * F, t2_b1 + (size_t)i * F,
        big, BT, F, Ed, t2_stan + i, nullptr);
    gemm_kernel<0, 0><<<dim3(Ed / 64, BT / 128), 256, 0, stream>>>(
        big, t2_w2 + (size_t)i * F * Ed, t2_b2 + (size_t)i * Ed,
        obuf, BT, Ed, F, nullptr, nullptr);
    resln_kernel<<<BT, 256, 0, stream>>>(hB, obuf, rs, t2_ln2g + (size_t)i * Ed,
                                         t2_ln2b + (size_t)i * Ed, Ed);
  }

  // 5) harmonic (Mahalanobis) softmax head
  var_kernel<<<256, 256, 0, stream>>>(hB, invb);
  x2_kernel<<<BT, 256, 0, stream>>>(hB, invb, x2b);
  w2_kernel<<<65, 256, 0, stream>>>(unemb, invb, w2b);
  // cross = h2 @ (unemb * inv[:,None]) : (BT,65), K=256, W stored (K,N)
  gemm_kernel<0, 0><<<dim3(2, BT / 128), 256, 0, stream>>>(
      hB, unemb, nullptr, big, BT, 65, 256, nullptr, invb);
  zero_kernel<<<1, 64, 0, stream>>>(dsum, 1);
  dist_kernel<<<(BT * 65 + 255) / 256, 256, 0, stream>>>(big, x2b, w2b, dbuf, dsum);
  headsm_kernel<<<BT, 128, 0, stream>>>(dbuf, dsum, out);
}
